// PraxisMemory_4578435137552
// MI455X (gfx1250) — matrix operand order repaired and checked by reading, not compiled, not run on hardware
//
#include <hip/hip_runtime.h>
#include <hip/hip_bf16.h>

typedef __attribute__((ext_vector_type(16))) __bf16 v16bf;
typedef __attribute__((ext_vector_type(8)))  float  v8f;

#define DD      1024
#define VV      32000
#define VPAD    32256
#define NCOLIT  504         /* 32256 / 64 cols per iteration */
#define CSPLIT  16
#define CPS     32          /* ceil(504/16) */

/* ---- output layout (flat f32, return-order concat) ---- */
static constexpr size_t OUT_Q_OFF = 0;                       /* (2,1024,1024)  */
static constexpr size_t OUT_K_OFF = 2097152;                 /* (2,1184,1024)  */
static constexpr size_t OUT_V_OFF = OUT_K_OFF + 2424832;     /* (2,1184,1024)  */
static constexpr size_t OUT_M_OFF = OUT_V_OFF + 2424832;     /* (2,1184)       */

/* ---- workspace layout (bytes, all 16B aligned) ---- */
static constexpr size_t OFF_QBF    = 0;
static constexpr size_t OFF_WCBF   = OFF_QBF    + 2097152ull  * 2;  /* 32256x1024 bf16 */
static constexpr size_t OFF_WS1BF  = OFF_WCBF   + 33030144ull * 2;
static constexpr size_t OFF_WS2BF  = OFF_WS1BF  + 1048576ull  * 2;
static constexpr size_t OFF_S0     = OFF_WS2BF  + 1048576ull  * 2;
static constexpr size_t OFF_S1     = OFF_S0     + 2048ull * 4;
static constexpr size_t OFF_SURP   = OFF_S1     + 2048ull * 4;
static constexpr size_t OFF_BND    = OFF_SURP   + 2048ull * 4;
static constexpr size_t OFF_COMM   = OFF_BND    + 2048ull * 4;
static constexpr size_t OFF_RVEC   = OFF_COMM   + 2048ull * 4;
static constexpr size_t OFF_RSUM   = OFF_RVEC   + 2097152ull * 4;
static constexpr size_t OFF_DEG    = OFF_RSUM   + 2048ull * 4;
static constexpr size_t OFF_CVEC   = OFF_DEG    + 2048ull * 4;      /* 2x1025x1024 */
static constexpr size_t OFF_CDEG   = OFF_CVEC   + 2099200ull * 4;   /* 2x1025 (pad) */
static constexpr size_t OFF_QV     = OFF_CDEG   + 2112ull * 4;
static constexpr size_t OFF_LAB    = OFF_QV     + 64ull * 4;
static constexpr size_t OFF_POS    = OFF_LAB    + 2048ull * 4;
static constexpr size_t OFF_EVTS   = OFF_POS    + 2048ull * 4;      /* 256x32x1024 f32 */
static constexpr size_t OFF_QBAR   = OFF_EVTS   + 8388608ull * 4;
static constexpr size_t OFF_QM     = OFF_QBAR   + 2048ull * 4;
static constexpr size_t OFF_SCORES = OFF_QM     + 2048ull * 4;
static constexpr size_t OFF_TOP    = OFF_SCORES + 1024ull * 4;
static constexpr size_t OFF_XBF    = OFF_TOP    + 64ull * 4;        /* 320x1024 bf16 */
static constexpr size_t OFF_HBF    = OFF_XBF    + 327680ull * 2;

union BfFrag { uint4 u[2]; v16bf v; };

__device__ __forceinline__ v16bf ld_afrag(const __bf16* arow) {
  /* A 16x32 bf16 layout: e0..7 at +8*hi, e8..15 at +16+8*hi (hi folded into arow) */
  BfFrag f;
  f.u[0] = *(const uint4*)(arow);
  f.u[1] = *(const uint4*)(arow + 16);
  return f.v;
}
__device__ __forceinline__ v16bf ld_bfrag(const __bf16* brow) {
  /* B 32x16 bf16 layout: e0..7 at +8*hi, e8..15 at +16+8*hi (hi folded into brow) */
  BfFrag f;
  f.u[0] = *(const uint4*)(brow);
  f.u[1] = *(const uint4*)(brow + 16);
  return f.v;
}

/* ---------------- utility kernels ---------------- */
__global__ void k_cvt_bf16(const float* __restrict__ s, __bf16* __restrict__ d, long n) {
  long i = (long)blockIdx.x * blockDim.x + threadIdx.x;
  long st = (long)gridDim.x * blockDim.x;
  for (; i < n; i += st) d[i] = (__bf16)s[i];
}
__global__ void k_zero_f32(float* p, long n) {
  long i = (long)blockIdx.x * blockDim.x + threadIdx.x;
  long st = (long)gridDim.x * blockDim.x;
  for (; i < n; i += st) p[i] = 0.f;
}
__global__ void k_zero_bf16(__bf16* p, long n) {
  long i = (long)blockIdx.x * blockDim.x + threadIdx.x;
  long st = (long)gridDim.x * blockDim.x;
  for (; i < n; i += st) p[i] = (__bf16)0.0f;
}
__global__ void k_copy_f32(const float* __restrict__ s, float* __restrict__ d, long n) {
  long i = (long)blockIdx.x * blockDim.x + threadIdx.x;
  long st = (long)gridDim.x * blockDim.x;
  for (; i < n; i += st) d[i] = s[i];
}

/* ---------------- fused logits GEMM + streaming entropy ----------------
   128 rows per workgroup (8 waves x one 16-row M-tile each), A staged once
   in 256KB dynamic LDS. All 8 waves iterate the SAME 64-column tile so B
   fragment fetches coalesce in the WGP$ (8x L2 traffic reduction).
   k-loop is manually 2x unrolled with all 10 fragments in distinct live
   ranges so the scheduler can overlap loads with WMMAs (no full-drain
   s_wait_loadcnt 0x0 per WMMA, no WAR v_nops on the A fragment).
   Accumulates S0 = sum_v e^l , S1 = sum_v l*e^l per row. */
__global__ __launch_bounds__(256) void k_logits_entropy(
    const __bf16* __restrict__ qbf, const __bf16* __restrict__ wcbf,
    float* __restrict__ s0buf, float* __restrict__ s1buf)
{
  extern __shared__ __align__(16) char smem[];
  __bf16* At = (__bf16*)smem;                  /* 128 x 1024 bf16 = 256KB */
  __shared__ float ls0[128], ls1[128];
  const int tid = threadIdx.x, wave = tid >> 5, lane = tid & 31;
  const int lo = lane & 15, hi = lane >> 4;
  const int m0 = blockIdx.x * 128;

  { /* stage A: 128 rows x 1024 k, vectorized 16B copies */
    const uint4* qsrc = (const uint4*)(qbf + (size_t)m0 * DD);
    uint4* adst = (uint4*)At;
    for (int i = tid; i < 128 * DD / 8; i += 256) adst[i] = qsrc[i];
  }
  if (tid < 128) { ls0[tid] = 0.f; ls1[tid] = 0.f; }
  __syncthreads();

  float p0[8], p1[8];
#pragma unroll
  for (int j = 0; j < 8; ++j) { p0[j] = 0.f; p1[j] = 0.f; }

  const __bf16* abase = At + ((size_t)(wave * 16 + lo)) * DD + 8 * hi;

  const int c0 = blockIdx.y * CPS;
  const int c1 = (c0 + CPS < NCOLIT) ? (c0 + CPS) : NCOLIT;

  for (int c = c0; c < c1; ++c) {
    const int v0 = c * 64;                      /* same columns for all waves */
    const __bf16* bbase = wcbf + (size_t)(v0 + lo) * DD + 8 * hi;
    v8f acc[4];
#pragma unroll
    for (int nt = 0; nt < 4; ++nt) acc[nt] = (v8f)0.0f;
    for (int k0 = 0; k0 < DD; k0 += 64) {
      /* issue all 20 b128 loads for two k-steps up front */
      v16bf a0 = ld_afrag(abase + k0);
      v16bf a1 = ld_afrag(abase + k0 + 32);
      v16bf b0[4], b1[4];
#pragma unroll
      for (int nt = 0; nt < 4; ++nt) {
        const __bf16* br = bbase + (size_t)(nt * 16) * DD + k0;
        b0[nt] = ld_bfrag(br);
        b1[nt] = ld_bfrag(br + 32);
      }
#pragma unroll
      for (int nt = 0; nt < 4; ++nt)
        acc[nt] = __builtin_amdgcn_wmma_f32_16x16x32_bf16(
            false, a0, false, b0[nt], (short)0, acc[nt], false, false);
#pragma unroll
      for (int nt = 0; nt < 4; ++nt)
        acc[nt] = __builtin_amdgcn_wmma_f32_16x16x32_bf16(
            false, a1, false, b1[nt], (short)0, acc[nt], false, false);
    }
#pragma unroll
    for (int nt = 0; nt < 4; ++nt) {
      const int col = v0 + nt * 16 + lo;
      if (col < VV) {
#pragma unroll
        for (int j = 0; j < 8; ++j) {
          float l = acc[nt][j];
          float e = __expf(l);
          p0[j] += e;
          p1[j] += l * e;
        }
      }
    }
  }
#pragma unroll
  for (int j = 0; j < 8; ++j) {
    atomicAdd(&ls0[wave * 16 + hi * 8 + j], p0[j]);
    atomicAdd(&ls1[wave * 16 + hi * 8 + j], p1[j]);
  }
  __syncthreads();
  if (tid < 128) {
    atomicAdd(&s0buf[m0 + tid], ls0[tid]);
    atomicAdd(&s1buf[m0 + tid], ls1[tid]);
  }
}

__global__ void k_surprise(const float* __restrict__ s0, const float* __restrict__ s1,
                           float* __restrict__ s) {
  int i = blockIdx.x * blockDim.x + threadIdx.x;
  if (i < 2048) {
    float a = s0[i];
    s[i] = (__logf(a) - s1[i] / a) * 0.09639955f; /* 1/ln(32000) */
  }
}

/* ---------------- windowed boundary detection (serial per batch) ------- */
__global__ void k_boundaries(const float* __restrict__ s, float* __restrict__ bnd,
                             int* __restrict__ comm) {
  int b = threadIdx.x;
  if (b >= 2) return;
  const float* sb = s + b * 1024;
  float rcs[32], rcs2[32];
  float cs = 0.f, cs2 = 0.f;
  int cm = 0;
  for (int t = 0; t < 1024; ++t) {
    float x = sb[t];
    float csl  = (t >= 32) ? rcs[t & 31]  : 0.f;
    float cs2l = (t >= 32) ? rcs2[t & 31] : 0.f;
    cs  += x;
    cs2 += x * x;
    rcs[t & 31] = cs;
    rcs2[t & 31] = cs2;
    float cnt = fminf((float)(t + 1), 32.f);
    float mu  = (cs - csl) / cnt;
    float var = fmaxf((cs2 - cs2l) / cnt - mu * mu, 0.f);
    float thr = mu + 2.f * sqrtf(var);
    float bt = (x > thr && x > 0.5f) ? 1.f : 0.f;
    bnd[b * 1024 + t] = bt;
    cm += (int)bt;
    comm[b * 1024 + t] = cm;
  }
}

/* ---------------- r = normalize(q); accumulate R and per-community sums */
__global__ __launch_bounds__(256) void k_rnorm(const float* __restrict__ q,
    const int* __restrict__ comm, float* __restrict__ rvec,
    float* __restrict__ Rsum, float* __restrict__ cvec)
{
  __shared__ float red[256];
  const int row = blockIdx.x, b = row >> 10, tid = threadIdx.x;
  const float* qr = q + (size_t)row * DD;
  float ss = 0.f;
  for (int d = tid; d < DD; d += 256) { float v = qr[d]; ss += v * v; }
  red[tid] = ss; __syncthreads();
  for (int o = 128; o > 0; o >>= 1) { if (tid < o) red[tid] += red[tid + o]; __syncthreads(); }
  const float inv = 1.f / (sqrtf(red[0]) + 1e-8f);
  float* cv = cvec + ((size_t)b * 1025 + comm[row]) * DD;
  float* Rs = Rsum + (size_t)b * DD;
  float* rr = rvec + (size_t)row * DD;
  for (int d = tid; d < DD; d += 256) {
    float v = qr[d] * inv;
    rr[d] = v;
    atomicAdd(&Rs[d], v);
    atomicAdd(&cv[d], v);
  }
}

__global__ __launch_bounds__(256) void k_deg(const float* __restrict__ rvec,
    const float* __restrict__ Rsum, const int* __restrict__ comm,
    float* __restrict__ deg, float* __restrict__ cdeg)
{
  __shared__ float red[256];
  const int row = blockIdx.x, b = row >> 10, tid = threadIdx.x;
  const float* rr = rvec + (size_t)row * DD;
  const float* Rs = Rsum + (size_t)b * DD;
  float s = 0.f;
  for (int d = tid; d < DD; d += 256) s += rr[d] * Rs[d];
  red[tid] = s; __syncthreads();
  for (int o = 128; o > 0; o >>= 1) { if (tid < o) red[tid] += red[tid + o]; __syncthreads(); }
  if (tid == 0) {
    float dv = red[0];
    deg[row] = dv;
    atomicAdd(&cdeg[b * 1025 + comm[row]], dv);
  }
}

/* Q = (sum_c |sum r|^2 - sum_c (sum deg)^2 / tot) / tot ; tot = |R|^2 */
__global__ __launch_bounds__(256) void k_modQ(const float* __restrict__ Rsum,
    const float* __restrict__ cvec, const float* __restrict__ cdeg,
    float* __restrict__ Qv)
{
  __shared__ float red[256];
  const int b = blockIdx.x, tid = threadIdx.x;
  float t = 0.f;
  for (int d = tid; d < DD; d += 256) { float v = Rsum[b * DD + d]; t += v * v; }
  red[tid] = t; __syncthreads();
  for (int o = 128; o > 0; o >>= 1) { if (tid < o) red[tid] += red[tid + o]; __syncthreads(); }
  const float tot = red[0] + 1e-10f;
  __syncthreads();
  float s1 = 0.f;
  const float* cv = cvec + (size_t)b * 1025 * DD;
  for (size_t i = tid; i < (size_t)1025 * DD; i += 256) { float v = cv[i]; s1 += v * v; }
  red[tid] = s1; __syncthreads();
  for (int o = 128; o > 0; o >>= 1) { if (tid < o) red[tid] += red[tid + o]; __syncthreads(); }
  s1 = red[0];
  __syncthreads();
  float s2 = 0.f;
  for (int c = tid; c < 1025; c += 256) { float v = cdeg[b * 1025 + c]; s2 += v * v; }
  red[tid] = s2; __syncthreads();
  for (int o = 128; o > 0; o >>= 1) { if (tid < o) red[tid] += red[tid + o]; __syncthreads(); }
  if (tid == 0) Qv[b] = (s1 - red[0] / tot) / tot;
}

/* gated labels / positions (serial per batch) */
__global__ void k_label(const float* __restrict__ bnd, const float* __restrict__ Qv,
                        int* __restrict__ lab, int* __restrict__ pos) {
  int b = threadIdx.x;
  if (b >= 2) return;
  const float g = (Qv[b] > 0.f) ? 1.f : 0.f;
  int cum = 0, prev = -1, start = 0;
  for (int t = 0; t < 1024; ++t) {
    cum += (int)(bnd[b * 1024 + t] * g);
    int l = cum < 127 ? cum : 127;
    if (t == 0 || l != prev) start = t;
    pos[b * 1024 + t] = t - start;
    lab[b * 1024 + t] = l;
    prev = l;
  }
}

__global__ __launch_bounds__(256) void k_scatter(const float* __restrict__ q,
    const int* __restrict__ lab, const int* __restrict__ pos, float* __restrict__ evts)
{
  const int row = blockIdx.x, b = row >> 10;
  const int p = pos[row];
  if (p >= 32) return;
  float* dst = evts + (((size_t)(b * 128 + lab[row])) * 32 + p) * DD;
  const float* src = q + (size_t)row * DD;
  for (int d = threadIdx.x; d < DD; d += 256) dst[d] = src[d];
}

__global__ void k_qbar(const float* __restrict__ q, float* __restrict__ qbar) {
  int i = blockIdx.x * blockDim.x + threadIdx.x;
  if (i >= 2048) return;
  int b = i >> 10, d = i & 1023;
  const float* p = q + (size_t)b * 1024 * 1024 + d;
  float s = 0.f;
  for (int t = 0; t < 1024; ++t) s += p[(size_t)t * 1024];
  qbar[i] = s * (1.f / 1024.f);
}

__global__ __launch_bounds__(256) void k_qm(const float* __restrict__ qbar,
    const float* __restrict__ Wsim, const float* __restrict__ bsim, float* __restrict__ qm)
{
  __shared__ float red[256];
  const int o = blockIdx.x, b = o >> 10, i = o & 1023, tid = threadIdx.x;
  float s = 0.f;
  for (int d = tid; d < DD; d += 256) s += qbar[b * DD + d] * Wsim[(size_t)i * DD + d];
  red[tid] = s; __syncthreads();
  for (int w = 128; w > 0; w >>= 1) { if (tid < w) red[tid] += red[tid + w]; __syncthreads(); }
  if (tid == 0) qm[o] = red[0] + bsim[i];
}

__global__ __launch_bounds__(256) void k_scores(const float* __restrict__ qm,
    const float* __restrict__ evts, const float* __restrict__ memk, float* __restrict__ sc)
{
  __shared__ float red[256];
  const int o = blockIdx.x, b = o >> 9, m = o & 511, tid = threadIdx.x;
  const float* key = (m < 256) ? (evts + (size_t)m * 32 * DD) : (memk + (size_t)m * DD);
  float s = 0.f;
  for (int d = tid; d < DD; d += 256) s += qm[b * DD + d] * key[d];
  red[tid] = s; __syncthreads();
  for (int w = 128; w > 0; w >>= 1) { if (tid < w) red[tid] += red[tid + w]; __syncthreads(); }
  if (tid == 0) sc[o] = red[0];
}

__global__ void k_top3(const float* __restrict__ sc, int* __restrict__ top) {
  int b = threadIdx.x;
  if (b >= 2) return;
  const float* s = sc + b * 512;
  int u0 = -1, u1 = -1;
  for (int k = 0; k < 3; ++k) {
    float best = -3.4e38f; int bi = 0;
    for (int m = 0; m < 512; ++m) {
      if (m == u0 || m == u1) continue;
      float v = s[m];
      if (v > best) { best = v; bi = m; }   /* strict > : ties pick lowest index */
    }
    top[b * 3 + k] = bi;
    if (k == 0) u0 = bi; else if (k == 1) u1 = bi;
  }
}

/* gather sim_buf (3x32) + cont_buf (2x32) rows, convert to bf16 */
__global__ __launch_bounds__(256) void k_gatherX(const float* __restrict__ evts,
    const float* __restrict__ memv, const int* __restrict__ top, __bf16* __restrict__ xbf)
{
  const int g = blockIdx.x;          /* 0..319 */
  const int b = g / 160, s = g % 160;
  int row, off;
  if (s < 96) { row = top[b * 3 + s / 32]; off = s & 31; }
  else {
    int j = (s - 96) >> 5;
    int t1 = top[b * 3];
    row = (j == 0) ? ((t1 + 511) & 511) : ((t1 + 1) & 511);
    off = (s - 96) & 31;
  }
  const float* src = (row < 256) ? (evts + ((size_t)row * 32 + off) * DD)
                                 : (memv + ((size_t)row * 32 + off) * DD);
  __bf16* dst = xbf + (size_t)g * DD;
  for (int d = threadIdx.x; d < DD; d += 256) dst[d] = (__bf16)src[d];
}

/* storage MLP layer via WMMA: out[g,col] = X[g,:] @ W[col,:] + bias[col]
   mode 0: relu -> bf16 Hout ; mode 1: f32 -> key_ctx & val_ctx slots of d_out */
__global__ __launch_bounds__(256) void k_mlp(const __bf16* __restrict__ X,
    const __bf16* __restrict__ W, const float* __restrict__ bias,
    __bf16* __restrict__ Hout, float* __restrict__ outbase, int mode)
{
  __shared__ __align__(16) __bf16 At[16 * DD];
  const int tid = threadIdx.x, wave = tid >> 5, lane = tid & 31;
  const int lo = lane & 15, hi = lane >> 4;
  const int m0 = blockIdx.x * 16;
  for (int i = tid; i < 16 * DD; i += 256) At[i] = X[(size_t)m0 * DD + i];
  __syncthreads();
  const int n0 = blockIdx.y * 512 + wave * 64;
  const __bf16* abase = At + (size_t)lo * DD + 8 * hi;
  const __bf16* bbase = W + (size_t)(n0 + lo) * DD + 8 * hi;
  v8f acc[4];
#pragma unroll
  for (int nt = 0; nt < 4; ++nt) acc[nt] = (v8f)0.0f;
  for (int k0 = 0; k0 < DD; k0 += 64) {
    v16bf a0 = ld_afrag(abase + k0);
    v16bf a1 = ld_afrag(abase + k0 + 32);
    v16bf b0[4], b1[4];
#pragma unroll
    for (int nt = 0; nt < 4; ++nt) {
      const __bf16* br = bbase + (size_t)(nt * 16) * DD + k0;
      b0[nt] = ld_bfrag(br);
      b1[nt] = ld_bfrag(br + 32);
    }
#pragma unroll
    for (int nt = 0; nt < 4; ++nt)
      acc[nt] = __builtin_amdgcn_wmma_f32_16x16x32_bf16(
          false, a0, false, b0[nt], (short)0, acc[nt], false, false);
#pragma unroll
    for (int nt = 0; nt < 4; ++nt)
      acc[nt] = __builtin_amdgcn_wmma_f32_16x16x32_bf16(
          false, a1, false, b1[nt], (short)0, acc[nt], false, false);
  }
#pragma unroll
  for (int nt = 0; nt < 4; ++nt) {
    const int col = n0 + nt * 16 + lo;
    const float bv = bias[col];
#pragma unroll
    for (int j = 0; j < 8; ++j) {
      const int g = m0 + j + 8 * hi;
      float v = acc[nt][j] + bv;
      if (mode == 0) {
        Hout[(size_t)g * DD + col] = (__bf16)fmaxf(v, 0.f);
      } else {
        const int b2 = g / 160, s2 = g % 160;
        size_t ko = OUT_K_OFF + ((size_t)(b2 * 1184 + s2)) * DD + col;
        outbase[ko] = v;
        outbase[ko + (OUT_V_OFF - OUT_K_OFF)] = v;
      }
    }
  }
}

/* key_in / value_in -> ctx rows 160..1183 */
__global__ void k_copy_ctx(const float* __restrict__ src, float* __restrict__ out, size_t base) {
  long i = (long)blockIdx.x * blockDim.x + threadIdx.x;
  long st = (long)gridDim.x * blockDim.x;
  for (; i < 2097152; i += st) {
    int b = (int)(i >> 20), s = (int)((i >> 10) & 1023), d = (int)(i & 1023);
    out[base + ((size_t)(b * 1184 + 160 + s)) * DD + d] = src[i];
  }
}

__global__ void k_mask(const float* __restrict__ am, float* __restrict__ out) {
  int i = blockIdx.x * blockDim.x + threadIdx.x;
  if (i >= 2368) return;
  int b = i / 1184, s = i % 1184;
  out[OUT_M_OFF + i] = (s < 160) ? 1.f : am[b * 1024 + (s - 160)];
}

extern "C" void kernel_launch(void* const* d_in, const int* in_sizes, int n_in,
                              void* d_out, int out_size, void* d_ws, size_t ws_size,
                              hipStream_t stream) {
  const float* q    = (const float*)d_in[0];
  const float* kin  = (const float*)d_in[1];
  const float* vin  = (const float*)d_in[2];
  const float* am   = (const float*)d_in[3];
  const float* Wc   = (const float*)d_in[4];
  const float* Wsim = (const float*)d_in[5];
  const float* bsim = (const float*)d_in[6];
  const float* Ws1  = (const float*)d_in[7];
  const float* bs1  = (const float*)d_in[8];
  const float* Ws2  = (const float*)d_in[9];
  const float* bs2  = (const float*)d_in[10];
  const float* memk = (const float*)d_in[11];
  const float* memv = (const float*)d_in[12];
  float* out = (float*)d_out;
  char*  ws  = (char*)d_ws;

  __bf16* qbf   = (__bf16*)(ws + OFF_QBF);
  __bf16* wcbf  = (__bf16*)(ws + OFF_WCBF);
  __bf16* ws1bf = (__bf16*)(ws + OFF_WS1BF);
  __bf16* ws2bf = (__bf16*)(ws + OFF_WS2BF);
  float* s0   = (float*)(ws + OFF_S0);
  float* s1   = (float*)(ws + OFF_S1);
  float* surp = (float*)(ws + OFF_SURP);
  float* bnd  = (float*)(ws + OFF_BND);
  int*   comm = (int*)(ws + OFF_COMM);
  float* rvec = (float*)(ws + OFF_RVEC);
  float* Rsum = (float*)(ws + OFF_RSUM);
  float* deg  = (float*)(ws + OFF_DEG);
  float* cvec = (float*)(ws + OFF_CVEC);
  float* cdeg = (float*)(ws + OFF_CDEG);
  float* Qv   = (float*)(ws + OFF_QV);
  int*   lab  = (int*)(ws + OFF_LAB);
  int*   pos  = (int*)(ws + OFF_POS);
  float* evts = (float*)(ws + OFF_EVTS);
  float* qbar = (float*)(ws + OFF_QBAR);
  float* qm   = (float*)(ws + OFF_QM);
  float* sc   = (float*)(ws + OFF_SCORES);
  int*   top  = (int*)(ws + OFF_TOP);
  __bf16* xbf = (__bf16*)(ws + OFF_XBF);
  __bf16* hbf = (__bf16*)(ws + OFF_HBF);

  /* conversions + padding */
  k_cvt_bf16<<<4096, 256, 0, stream>>>(q, qbf, 2097152L);
  k_cvt_bf16<<<8192, 256, 0, stream>>>(Wc, wcbf, 32768000L);
  k_zero_bf16<<<512, 256, 0, stream>>>(wcbf + 32768000L, 262144L);
  k_cvt_bf16<<<2048, 256, 0, stream>>>(Ws1, ws1bf, 1048576L);
  k_cvt_bf16<<<2048, 256, 0, stream>>>(Ws2, ws2bf, 1048576L);

  /* zero accumulators */
  k_zero_f32<<<16, 256, 0, stream>>>(s0, 4096L);        /* s0 + s1 contiguous */
  k_zero_f32<<<8, 256, 0, stream>>>(Rsum, 2048L);
  k_zero_f32<<<2048, 256, 0, stream>>>(cvec, 2099200L);
  k_zero_f32<<<9, 256, 0, stream>>>(cdeg, 2112L);
  k_zero_f32<<<4096, 256, 0, stream>>>(evts, 8388608L);

  /* fused logits GEMM + streaming entropy: 16 row-blocks x 16 col-splits,
     256KB dynamic LDS for the 128-row A tile */
  k_logits_entropy<<<dim3(16, CSPLIT), 256, 128 * DD * sizeof(__bf16), stream>>>(
      qbf, wcbf, s0, s1);
  k_surprise<<<8, 256, 0, stream>>>(s0, s1, surp);

  /* boundaries + modularity (implicit, no SxS sim materialization) */
  k_boundaries<<<1, 32, 0, stream>>>(surp, bnd, comm);
  k_rnorm<<<2048, 256, 0, stream>>>(q, comm, rvec, Rsum, cvec);
  k_deg<<<2048, 256, 0, stream>>>(rvec, Rsum, comm, deg, cdeg);
  k_modQ<<<2, 256, 0, stream>>>(Rsum, cvec, cdeg, Qv);

  /* event segmentation + memory update */
  k_label<<<1, 32, 0, stream>>>(bnd, Qv, lab, pos);
  k_scatter<<<2048, 256, 0, stream>>>(q, lab, pos, evts);

  /* retrieval */
  k_qbar<<<8, 256, 0, stream>>>(q, qbar);
  k_qm<<<2048, 256, 0, stream>>>(qbar, Wsim, bsim, qm);
  k_scores<<<1024, 256, 0, stream>>>(qm, evts, memk, sc);
  k_top3<<<1, 32, 0, stream>>>(sc, top);
  k_gatherX<<<320, 256, 0, stream>>>(evts, memv, top, xbf);

  /* storage MLP (WMMA), layer2 writes straight into key_ctx & val_ctx */
  k_mlp<<<dim3(20, 2), 256, 0, stream>>>(xbf, ws1bf, bs1, hbf, nullptr, 0);
  k_mlp<<<dim3(20, 2), 256, 0, stream>>>(hbf, ws2bf, bs2, nullptr, out, 1);

  /* passthrough outputs */
  k_copy_f32<<<4096, 256, 0, stream>>>(q, out + OUT_Q_OFF, 2097152L);
  k_copy_ctx<<<4096, 256, 0, stream>>>(kin, out, OUT_K_OFF);
  k_copy_ctx<<<4096, 256, 0, stream>>>(vin, out, OUT_V_OFF);
  k_mask<<<10, 256, 0, stream>>>(am, out);
  (void)in_sizes; (void)n_in; (void)out_size; (void)ws_size;
}
